// SOMLayer_3513283248278
// MI455X (gfx1250) — matrix operand order repaired and checked by reading, not compiled, not run on hardware
//
#include <hip/hip_runtime.h>
#include <hip/hip_bf16.h>

// SOM layer: out[b] = mean_p ||x_b - proto_p||^2
//          = ||x_b||^2 + mean_p||p||^2 - (2/P) * sum_p x_b . proto_p
// Cross term via v_wmma_f32_16x16x32_bf16, mean folded into the accumulator.

typedef __attribute__((ext_vector_type(16))) __bf16 v16bf;
typedef __attribute__((ext_vector_type(8)))  float  v8f;

#define B_SZ 8192
#define P_SZ 1024
#define D_SZ 256

// ---- pre-pass 1: prototypes fp32 -> bf16 (done once per launch, 512 KB in ws)
__global__ void som_cvt_proto(const float* __restrict__ p, __bf16* __restrict__ pbf, int n) {
    int i = blockIdx.x * blockDim.x + threadIdx.x;
    if (i < n) pbf[i] = (__bf16)p[i];
}

// ---- pre-pass 2: c0 = (1/P) * sum_{p,d} proto^2  (single block -> deterministic)
__global__ void som_p2_mean(const float* __restrict__ p, float* __restrict__ c0) {
    __shared__ float sm[256];
    float s = 0.f;
    for (int i = threadIdx.x; i < P_SZ * D_SZ; i += 256) {
        float v = p[i];
        s += v * v;
    }
    sm[threadIdx.x] = s;
    __syncthreads();
    for (int st = 128; st > 0; st >>= 1) {
        if (threadIdx.x < st) sm[threadIdx.x] += sm[threadIdx.x + st];
        __syncthreads();
    }
    if (threadIdx.x == 0) c0[0] = sm[0] * (1.0f / (float)P_SZ);
}

// ---- main kernel: 4 waves/block, each wave owns a 16-row batch tile
__global__ __launch_bounds__(128) void som_wmma_kernel(
    const float* __restrict__ x, const __bf16* __restrict__ pbf,
    const float* __restrict__ c0p, float* __restrict__ out) {
    const int lane = threadIdx.x & 31;
    const int wave = threadIdx.x >> 5;
    const int half = lane >> 4;   // which 16-lane half of the wave32
    const int l16  = lane & 15;
    const int b0   = blockIdx.x * 64 + wave * 16;

    const float* xrow = x + (size_t)(b0 + l16) * D_SZ;

    // Preload + convert all 8 A fragments (K=0..255) for this wave's 16 rows.
    // A-layout (16x32 bf16): lane half 0 -> K = kk*32+{0..15},
    //                        lane half 1 -> K = kk*32+{16..31}.
    v16bf afrag[8];
    float x2p = 0.f;  // this lane covers exactly half of its row's elements
#pragma unroll
    for (int kk = 0; kk < 8; ++kk) {
        const int d0 = kk * 32 + half * 16;
        float f[16];
        *(float4*)(f + 0)  = *(const float4*)(xrow + d0 + 0);
        *(float4*)(f + 4)  = *(const float4*)(xrow + d0 + 4);
        *(float4*)(f + 8)  = *(const float4*)(xrow + d0 + 8);
        *(float4*)(f + 12) = *(const float4*)(xrow + d0 + 12);
        v16bf a;
#pragma unroll
        for (int e = 0; e < 16; ++e) {
            x2p += f[e] * f[e];
            a[e] = (__bf16)f[e];
        }
        afrag[kk] = a;
    }

    // B-layout (32x16 bf16): lane holds proto[p0 + l16][d0 + 16*half + {0..15}]
    const __bf16* pb = pbf + (size_t)l16 * D_SZ + half * 16;

    auto tile = [&](int pt, v8f c) -> v8f {
        const __bf16* brow = pb + (size_t)pt * (16 * D_SZ);
#pragma unroll
        for (int kk = 0; kk < 8; ++kk) {
            v16bf b = *(const v16bf*)(brow + kk * 32);
            c = __builtin_amdgcn_wmma_f32_16x16x32_bf16(
                    /*neg_a=*/false, afrag[kk], /*neg_b=*/false, b,
                    /*c_mod=*/(short)0, c, /*reuse_a=*/false, /*reuse_b=*/false);
        }
        return c;
    };

    // Accumulate C elementwise over all 64 p-tiles (mean over p folds into C);
    // 4 rotating accumulators give independent WMMA chains for overlap.
    v8f acc0 = {}, acc1 = {}, acc2 = {}, acc3 = {};
    for (int t = 0; t < 64; t += 4) {
        acc0 = tile(t + 0, acc0);
        acc1 = tile(t + 1, acc1);
        acc2 = tile(t + 2, acc2);
        acc3 = tile(t + 3, acc3);
    }
    v8f accT = (acc0 + acc1) + (acc2 + acc3);

    // Full ||x_b||^2: lane L and L+16 cover complementary halves of row b0+L%16
    const float x2row = x2p + __shfl_xor(x2p, 16, 32);
    const float c0v   = *c0p;
    const float scale = -2.0f / (float)P_SZ;

    // C-layout: VGPR r -> (M=r, N=lane) for lanes 0-15, (M=r+8, N=lane-16) for 16-31.
    // Row-sum over N = xor-reduce within each 16-lane half.
#pragma unroll
    for (int r = 0; r < 8; ++r) {
        float s = accT[r];
        s += __shfl_xor(s, 1, 32);
        s += __shfl_xor(s, 2, 32);
        s += __shfl_xor(s, 4, 32);
        s += __shfl_xor(s, 8, 32);
        const float xa = __shfl(x2row, r, 32);      // x2 of row r (held at lane r)
        const float xb = __shfl(x2row, r + 8, 32);  // x2 of row r+8
        if (lane == 0)       out[b0 + r]     = xa + c0v + scale * s;
        else if (lane == 16) out[b0 + 8 + r] = xb + c0v + scale * s;
    }
}

extern "C" void kernel_launch(void* const* d_in, const int* in_sizes, int n_in,
                              void* d_out, int out_size, void* d_ws, size_t ws_size,
                              hipStream_t stream) {
    (void)in_sizes; (void)n_in; (void)out_size; (void)ws_size;
    const float* x     = (const float*)d_in[0];   // (B, D) fp32
    const float* proto = (const float*)d_in[1];   // (P, D) fp32
    float*       out   = (float*)d_out;           // (B,)   fp32

    __bf16* pbf = (__bf16*)d_ws;                                  // 512 KB
    float*  c0  = (float*)((char*)d_ws + (size_t)P_SZ * D_SZ * 2); // 4 B

    som_cvt_proto<<<(P_SZ * D_SZ + 255) / 256, 256, 0, stream>>>(proto, pbf, P_SZ * D_SZ);
    som_p2_mean<<<1, 256, 0, stream>>>(proto, c0);
    som_wmma_kernel<<<B_SZ / 64, 128, 0, stream>>>(x, pbf, c0, out);
}
